// CNN_Decoder_55267639165378
// MI455X (gfx1250) — matrix operand order repaired and checked by reading, compiled, not run on hardware
//
#include <hip/hip_runtime.h>

typedef __attribute__((ext_vector_type(16))) __bf16 v16bf;
typedef __attribute__((ext_vector_type(8)))  __bf16 v8bf;
typedef __attribute__((ext_vector_type(8)))  float  v8f;

namespace {
constexpr int   kB    = 1024;
constexpr int   kD    = 200;
constexpr int   kOC   = 96;
constexpr int   kEnt  = 40000;
constexpr int   kHsp  = 20;
constexpr int   kWsp  = 20;
constexpr int   kSpat = 400;
constexpr int   kKA   = 115200;      // FC reduction dim = 3*96*400
constexpr int   kNFC  = 224;         // FC out cols padded to 14 tiles
constexpr int   kNTfc = 14;          // n-tiles in FC GEMM
constexpr int   kNTgrp = 7;          // n-tiles per wave (56 acc VGPRs)
constexpr int   kNG   = kNTfc / kNTgrp;    // 2 n-groups
constexpr int   kKTfc = kKA / 32;    // 3600
constexpr int   kKSpl = 16;          // K-split factor for FC GEMM
constexpr int   kKTchunk = kKTfc / kKSpl;  // 225 k-tiles per wave
constexpr int   kKH   = 224;         // scoring K padded to 7 tiles
constexpr int   kNTsc = kEnt / 16;   // 2500
constexpr int   kKTsc = kKH / 32;    // 7
constexpr float kEps  = 1e-5f;

// workspace layout (bytes)
constexpr size_t OFF_STATS = 0;                                        // 1024 f32
constexpr size_t OFF_XPRE  = 4096;                                     // B*400 f32
constexpr size_t OFF_POOL  = OFF_XPRE + (size_t)kB * kSpat * 4;        // B*3*96 f32 (spatial sums)
constexpr size_t OFF_CHSC  = OFF_POOL + (size_t)kB * 3 * kOC * 4;      // 288 scale + 288 shift f32
constexpr size_t OFF_PA    = OFF_CHSC + (size_t)3 * kOC * 2 * 4;       // B*3*96 f32
constexpr size_t OFF_PB    = OFF_PA + (size_t)kB * 3 * kOC * 4;        // B*3*96 f32
constexpr size_t OFF_A     = ((OFF_PB + (size_t)kB * 3 * kOC * 4) + 255) & ~(size_t)255; // B*115200 bf16
constexpr size_t OFF_WPK   = OFF_A + (size_t)kB * kKA * 2;             // 115200*224 bf16 packed tiles
constexpr size_t OFF_OFC   = OFF_WPK + (size_t)kKA * kNFC * 2;         // B*224 f32 (atomic accumulators)
constexpr size_t OFF_HBF   = OFF_OFC + (size_t)kB * kNFC * 4;          // B*224 bf16
constexpr size_t OFF_NFPK  = OFF_HBF + (size_t)kB * kKH * 2;           // 224*40000 bf16 packed tiles
} // namespace

// ---------------- small utility kernels ----------------

__global__ void k_zero_f32(float* p, int n) {
  int i = blockIdx.x * blockDim.x + threadIdx.x;
  if (i < n) p[i] = 0.f;
}

// gather + chequer reshape + BN0 global stats
__global__ void k_gather(const float* __restrict__ nf, const float* __restrict__ rf,
                         const int* __restrict__ sub, const int* __restrict__ rel,
                         float* __restrict__ xpre, float* __restrict__ stats) {
  __shared__ float red[128];
  int b = blockIdx.x, t = threadIdx.x;
  int sb = sub[b], rb = rel[b];
  float s = 0.f, q = 0.f;
  for (int p = t; p < kSpat; p += blockDim.x) {
    int h = p / kWsp, w = p % kWsp;
    int idx = h * 10 + (w >> 1);
    float v = (w & 1) ? rf[(size_t)rb * kD + idx] : nf[(size_t)sb * kD + idx];
    xpre[(size_t)b * kSpat + p] = v;
    s += v; q += v * v;
  }
  red[t] = s; __syncthreads();
  for (int st = 64; st > 0; st >>= 1) { if (t < st) red[t] += red[t + st]; __syncthreads(); }
  if (t == 0) atomicAdd(&stats[0], red[0]);
  __syncthreads();
  red[t] = q; __syncthreads();
  for (int st = 64; st > 0; st >>= 1) { if (t < st) red[t] += red[t + st]; __syncthreads(); }
  if (t == 0) atomicAdd(&stats[1], red[0]);
}

// per-sample dynamic 'same'-padded cross-correlation, single input channel
__device__ __forceinline__ float conv_pt(const float* __restrict__ xs,
                                         const float* __restrict__ f,
                                         int br, int h, int w) {
  float acc = 0.f;
  if (br == 0) {                       // 1x5, pad w by 2
    #pragma unroll
    for (int j = 0; j < 5; ++j) { int ww = w + j - 2; if (ww >= 0 && ww < kWsp) acc += xs[h * kWsp + ww] * f[j]; }
  } else if (br == 1) {                // 3x3, pad 1
    #pragma unroll
    for (int i = 0; i < 3; ++i) {
      int hh = h + i - 1; if (hh < 0 || hh >= kHsp) continue;
      #pragma unroll
      for (int j = 0; j < 3; ++j) { int ww = w + j - 1; if (ww >= 0 && ww < kWsp) acc += xs[hh * kWsp + ww] * f[i * 3 + j]; }
    }
  } else {                             // 5x1, pad h by 2
    #pragma unroll
    for (int i = 0; i < 5; ++i) { int hh = h + i - 2; if (hh >= 0 && hh < kHsp) acc += xs[hh * kWsp + w] * f[i]; }
  }
  return acc;
}

// conv pass 1: channel BN stats (atomics) + per-sample spatial pool sums
__global__ void k_conv_stats(const float* __restrict__ xpre, float* __restrict__ stats,
                             float* __restrict__ pool,
                             const float* __restrict__ f1, const float* __restrict__ f2,
                             const float* __restrict__ f3, const int* __restrict__ rel,
                             const float* __restrict__ bn0g, const float* __restrict__ bn0b) {
  __shared__ float xs[kSpat];
  __shared__ float ff[kOC * 9];
  __shared__ float lsum[kOC], lsq[kOC];
  int b = blockIdx.x / 3, k = blockIdx.x % 3, t = threadIdx.x;
  float n = (float)kB * kSpat;
  float mean = stats[0] / n;
  float var = stats[1] / n - mean * mean;
  float a = bn0g[0] * rsqrtf(var + kEps);
  float bb = bn0b[0] - mean * a;
  for (int i = t; i < kSpat; i += blockDim.x) xs[i] = a * xpre[(size_t)b * kSpat + i] + bb;
  const float* ftab = (k == 0) ? f1 : (k == 1) ? f2 : f3;
  int fl = (k == 1) ? 9 : 5;
  const float* frow = ftab + (size_t)rel[b] * (kOC * fl);
  for (int i = t; i < kOC * fl; i += blockDim.x) ff[i] = frow[i];
  if (t < kOC) { lsum[t] = 0.f; lsq[t] = 0.f; }
  __syncthreads();
  for (int idx = t; idx < kOC * kSpat; idx += blockDim.x) {
    int c = idx / kSpat, p = idx % kSpat;
    float y = conv_pt(xs, ff + c * fl, k, p / kWsp, p % kWsp);
    atomicAdd(&lsum[c], y);
    atomicAdd(&lsq[c], y * y);
  }
  __syncthreads();
  if (t < kOC) {
    pool[((size_t)b * 3 + k) * kOC + t] = lsum[t];
    atomicAdd(&stats[2 + k * kOC + t], lsum[t]);
    atomicAdd(&stats[290 + k * kOC + t], lsq[t]);
  }
}

// finalize per-channel BN scale/shift for the three branches
__global__ void k_chfin(const float* __restrict__ stats,
                        const float* g1, const float* b1, const float* g2, const float* b2,
                        const float* g3, const float* b3, float* __restrict__ chsc) {
  int i = blockIdx.x * blockDim.x + threadIdx.x;
  if (i >= 3 * kOC) return;
  int k = i / kOC, c = i % kOC;
  float n = (float)kB * kSpat;
  float mean = stats[2 + i] / n;
  float var = stats[290 + i] / n - mean * mean;
  const float* g = (k == 0) ? g1 : (k == 1) ? g2 : g3;
  const float* bb = (k == 0) ? b1 : (k == 1) ? b2 : b3;
  float a = g[c] * rsqrtf(var + kEps);
  chsc[i] = a;
  chsc[3 * kOC + i] = bb[c] - mean * a;
}

// SE gates + ATT softmax gates -> fused per-(b,branch,channel) affine coefficients
__global__ void k_gates(const float* __restrict__ pool, const float* __restrict__ chsc,
                        float* __restrict__ pA, float* __restrict__ pB,
                        const float* s1w1, const float* s1w2, const float* s2w1, const float* s2w2,
                        const float* s3w1, const float* s3w2, const float* aw1, const float* aw2) {
  __shared__ float zp[3][kOC];
  __shared__ float sg[3][kOC];
  __shared__ float px[kOC];
  __shared__ float red[128];
  __shared__ float hid[2];
  __shared__ float scal[2];
  int b = blockIdx.x, t = threadIdx.x;
  bool act = t < kOC;
  const float* w1s[3] = { s1w1, s2w1, s3w1 };
  const float* w2s[3] = { s1w2, s2w2, s3w2 };
  for (int k = 0; k < 3; ++k)
    if (act) {
      float ymean = pool[((size_t)b * 3 + k) * kOC + t] * (1.f / kSpat);
      zp[k][t] = chsc[k * kOC + t] * ymean + chsc[3 * kOC + k * kOC + t];
    }
  __syncthreads();
  for (int k = 0; k < 3; ++k) {
    for (int j = 0; j < 2; ++j) {
      red[t] = act ? w1s[k][j * kOC + t] * zp[k][t] : 0.f; __syncthreads();
      for (int st = 64; st > 0; st >>= 1) { if (t < st) red[t] += red[t + st]; __syncthreads(); }
      if (t == 0) hid[j] = fmaxf(red[0], 0.f);
      __syncthreads();
    }
    if (act) {
      float u = w2s[k][t * 2 + 0] * hid[0] + w2s[k][t * 2 + 1] * hid[1];
      sg[k][t] = 1.f / (1.f + __expf(-u));
    }
    __syncthreads();
  }
  if (act) px[t] = sg[0][t] * zp[0][t] + sg[1][t] * zp[1][t] + sg[2][t] * zp[2][t];
  __syncthreads();
  for (int j = 0; j < 2; ++j) {
    red[t] = act ? aw1[j * kOC + t] * px[t] : 0.f; __syncthreads();
    for (int st = 64; st > 0; st >>= 1) { if (t < st) red[t] += red[t + st]; __syncthreads(); }
    if (t == 0) hid[j] = fmaxf(red[0], 0.f);
    __syncthreads();
  }
  float l[3]; float lmax = -1e30f;
  for (int k = 0; k < 3; ++k) {
    l[k] = act ? aw2[(k * kOC + t) * 2 + 0] * hid[0] + aw2[(k * kOC + t) * 2 + 1] * hid[1] : -1e30f;
    lmax = fmaxf(lmax, l[k]);
  }
  red[t] = lmax; __syncthreads();
  for (int st = 64; st > 0; st >>= 1) { if (t < st) red[t] = fmaxf(red[t], red[t + st]); __syncthreads(); }
  if (t == 0) scal[0] = red[0];
  __syncthreads();
  float m = scal[0];
  float e[3]; float es = 0.f;
  for (int k = 0; k < 3; ++k) { e[k] = act ? __expf(l[k] - m) : 0.f; es += e[k]; }
  red[t] = es; __syncthreads();
  for (int st = 64; st > 0; st >>= 1) { if (t < st) red[t] += red[t + st]; __syncthreads(); }
  if (t == 0) scal[1] = red[0];
  __syncthreads();
  float inv = 1.f / scal[1];
  if (act)
    for (int k = 0; k < 3; ++k) {
      float G = sg[k][t] * e[k] * inv;            // sigmoid * softmax, > 0
      pA[((size_t)b * 3 + k) * kOC + t] = chsc[k * kOC + t] * G;
      pB[((size_t)b * 3 + k) * kOC + t] = chsc[3 * kOC + k * kOC + t] * G;
    }
}

// conv pass 2: recompute conv, apply fused BN*SE*ATT + relu, emit bf16 GEMM A-matrix row
__global__ void k_buildA(const float* __restrict__ xpre, const float* __restrict__ stats,
                         const float* __restrict__ f1, const float* __restrict__ f2,
                         const float* __restrict__ f3, const int* __restrict__ rel,
                         const float* __restrict__ bn0g, const float* __restrict__ bn0b,
                         const float* __restrict__ pA, const float* __restrict__ pB,
                         __bf16* __restrict__ Am) {
  __shared__ float xs[kSpat];
  __shared__ float ff[kOC * 9];
  __shared__ float pa[kOC], pb[kOC];
  int b = blockIdx.x / 3, k = blockIdx.x % 3, t = threadIdx.x;
  float n = (float)kB * kSpat;
  float mean = stats[0] / n;
  float var = stats[1] / n - mean * mean;
  float a = bn0g[0] * rsqrtf(var + kEps);
  float bb = bn0b[0] - mean * a;
  for (int i = t; i < kSpat; i += blockDim.x) xs[i] = a * xpre[(size_t)b * kSpat + i] + bb;
  const float* ftab = (k == 0) ? f1 : (k == 1) ? f2 : f3;
  int fl = (k == 1) ? 9 : 5;
  const float* frow = ftab + (size_t)rel[b] * (kOC * fl);
  for (int i = t; i < kOC * fl; i += blockDim.x) ff[i] = frow[i];
  if (t < kOC) { pa[t] = pA[((size_t)b * 3 + k) * kOC + t]; pb[t] = pB[((size_t)b * 3 + k) * kOC + t]; }
  __syncthreads();
  __bf16* arow = Am + (size_t)b * kKA + (size_t)k * kOC * kSpat;
  for (int idx = t; idx < kOC * kSpat; idx += blockDim.x) {
    int c = idx / kSpat, p = idx % kSpat;
    float y = conv_pt(xs, ff + c * fl, k, p / kWsp, p % kWsp);
    arow[idx] = (__bf16)fmaxf(y * pa[c] + pb[c], 0.f);
  }
}

// pack fc_w^T into WMMA B-fragment tile order (K(e) = 16*half + e), bf16, N padded 200->224
__global__ void k_pack_fcw(const float* __restrict__ fcw, __bf16* __restrict__ wpk) {
  size_t i = (size_t)blockIdx.x * blockDim.x + threadIdx.x;
  int e = (int)(i & 15);
  int lane = (int)((i >> 4) & 31);
  size_t tile = i >> 9;
  int nt = (int)(tile % kNTfc);
  int kt = (int)(tile / kNTfc);
  int kk = kt * 32 + (lane >> 4) * 16 + e;
  int nn = nt * 16 + (lane & 15);
  float v = (nn < kD) ? fcw[(size_t)nn * kKA + kk] : 0.f;
  wpk[i] = (__bf16)v;
}

// pack n_feats^T into WMMA B-fragment tile order, K padded 200->224
__global__ void k_pack_nf(const float* __restrict__ nf, __bf16* __restrict__ npk) {
  size_t i = (size_t)blockIdx.x * blockDim.x + threadIdx.x;
  int e = (int)(i & 15);
  int lane = (int)((i >> 4) & 31);
  size_t tile = i >> 9;
  int nt = (int)(tile % kNTsc);
  int kt = (int)(tile / kNTsc);
  int kk = kt * 32 + (lane >> 4) * 16 + e;
  int nn = nt * 16 + (lane & 15);
  float v = (kk < kD) ? nf[(size_t)nn * kD + kk] : 0.f;
  npk[i] = (__bf16)v;
}

// A-fragment: this code's contiguous k order (same as packed B) — lane half h holds K in {h*16..h*16+15}
__device__ __forceinline__ v16bf load_afrag(const __bf16* __restrict__ rowp, int kb, int hh) {
  v8bf lo = *(const v8bf*)(rowp + kb + hh * 16);
  v8bf hi = *(const v8bf*)(rowp + kb + hh * 16 + 8);
  v16bf r;
  #pragma unroll
  for (int i = 0; i < 8; ++i) { r[i] = lo[i]; r[i + 8] = hi[i]; }
  return r;
}
__device__ __forceinline__ v16bf load_bfrag(const __bf16* __restrict__ p) {
  v8bf lo = *(const v8bf*)(p);
  v8bf hi = *(const v8bf*)(p + 8);
  v16bf r;
  #pragma unroll
  for (int i = 0; i < 8; ++i) { r[i] = lo[i]; r[i + 8] = hi[i]; }
  return r;
}

// FC GEMM, A-stationary K-split form:
//   wave = (m-tile, n-group of 7 tiles, K-chunk of 225 k-tiles).
//   7 v8f accumulators (56 VGPRs) + launch_bounds(128,1) -> no spills.
//   A (236 MB, > L2) is read exactly twice; Wpk (~52 MB) stays L2-resident.
//   Partials combined with global f32 atomics into zero-initialized outfc.
__global__ void __launch_bounds__(128, 1)
k_fc_gemm(const __bf16* __restrict__ A, const __bf16* __restrict__ Wpk,
          float* __restrict__ outfc) {
  int t = threadIdx.x;
  int lane = t & 31;
  int tile = blockIdx.x * (blockDim.x >> 5) + (t >> 5);   // 0..2047
  int mt = tile / (kNG * kKSpl);
  int rem = tile % (kNG * kKSpl);
  int ng = rem / kKSpl;                                   // n-group: 0 or 1
  int kc = rem % kKSpl;
  int m = lane & 15, hh = lane >> 4;
  const __bf16* arow = A + (size_t)(mt * 16 + m) * kKA;
  v8f acc[kNTgrp];
  #pragma unroll
  for (int j = 0; j < kNTgrp; ++j) acc[j] = (v8f){};
  int kt0 = kc * kKTchunk;
  int nt0 = ng * kNTgrp;
  for (int kt = kt0; kt < kt0 + kKTchunk; ++kt) {
    v16bf af = load_afrag(arow, kt * 32, hh);
    const __bf16* bp0 = Wpk + (((size_t)kt * kNTfc + nt0) * 32 + lane) * 16;
    __builtin_prefetch(bp0 + (size_t)kNTfc * 512, 0, 1);  // next kt's B tiles (L2-resident)
    #pragma unroll
    for (int j = 0; j < kNTgrp; ++j) {
      v16bf bf_ = load_bfrag(bp0 + (size_t)j * 512);
      acc[j] = __builtin_amdgcn_wmma_f32_16x16x32_bf16(false, af, false, bf_, (short)0, acc[j], false, false);
    }
  }
  #pragma unroll
  for (int j = 0; j < kNTgrp; ++j)
    #pragma unroll
    for (int r = 0; r < 8; ++r)
      atomicAdd(&outfc[(size_t)(mt * 16 + r + 8 * hh) * kNFC + (nt0 + j) * 16 + m], acc[j][r]);
}

// BatchNorm1d over batch (200 features) + relu -> bf16 h (B x 224, zero-padded)
__global__ void k_bn2(const float* __restrict__ outfc, const float* __restrict__ g,
                      const float* __restrict__ bvec, __bf16* __restrict__ hbf) {
  __shared__ float red[256];
  int d = blockIdx.x, t = threadIdx.x;
  if (d >= kD) {
    for (int b = t; b < kB; b += blockDim.x) hbf[(size_t)b * kKH + d] = (__bf16)0.f;
    return;
  }
  float s = 0.f, q = 0.f;
  for (int b = t; b < kB; b += blockDim.x) {
    float v = outfc[(size_t)b * kNFC + d];
    s += v; q += v * v;
  }
  red[t] = s; __syncthreads();
  for (int st = 128; st > 0; st >>= 1) { if (t < st) red[t] += red[t + st]; __syncthreads(); }
  float mean = red[0] / kB; __syncthreads();
  red[t] = q; __syncthreads();
  for (int st = 128; st > 0; st >>= 1) { if (t < st) red[t] += red[t + st]; __syncthreads(); }
  float var = red[0] / kB - mean * mean;
  float a = g[d] * rsqrtf(var + kEps);
  float bb = bvec[d] - mean * a;
  for (int b = t; b < kB; b += blockDim.x) {
    float v = a * outfc[(size_t)b * kNFC + d] + bb;
    hbf[(size_t)b * kKH + d] = (__bf16)fmaxf(v, 0.f);
  }
}

// scoring GEMM: (1024 x 224) bf16 @ packed (224 x 40000) bf16 + bias -> f32 scores
__global__ void k_score_gemm(const __bf16* __restrict__ Hm, const __bf16* __restrict__ Npk,
                             const float* __restrict__ bias, float* __restrict__ out) {
  int t = threadIdx.x;
  int lane = t & 31;
  int tile = blockIdx.x * (blockDim.x >> 5) + (t >> 5);
  int mt = tile / kNTsc, nt = tile % kNTsc;
  int m = lane & 15, hh = lane >> 4;
  const __bf16* arow = Hm + (size_t)(mt * 16 + m) * kKH;
  v8f acc = {};
  #pragma unroll
  for (int kt = 0; kt < kKTsc; ++kt) {
    v16bf af = load_afrag(arow, kt * 32, hh);
    const __bf16* bp = Npk + (((size_t)kt * kNTsc + nt) * 32 + lane) * 16;
    v16bf bf_ = load_bfrag(bp);
    acc = __builtin_amdgcn_wmma_f32_16x16x32_bf16(false, af, false, bf_, (short)0, acc, false, false);
  }
  int nn = nt * 16 + m;
  float bv = bias[nn];
  #pragma unroll
  for (int r = 0; r < 8; ++r)
    out[(size_t)(mt * 16 + r + 8 * hh) * kEnt + nn] = acc[r] + bv;
}

extern "C" void kernel_launch(void* const* d_in, const int* in_sizes, int n_in,
                              void* d_out, int out_size, void* d_ws, size_t ws_size,
                              hipStream_t stream) {
  const float* n_feats = (const float*)d_in[0];
  const float* r_feats = (const float*)d_in[1];
  const float* filt1   = (const float*)d_in[2];
  const float* filt2   = (const float*)d_in[3];
  const float* filt3   = (const float*)d_in[4];
  const float* bn0_g   = (const float*)d_in[5];
  const float* bn0_b   = (const float*)d_in[6];
  const float* bn1_g   = (const float*)d_in[7];
  const float* bn1_b   = (const float*)d_in[8];
  const float* bn2c_g  = (const float*)d_in[9];
  const float* bn2c_b  = (const float*)d_in[10];
  const float* bn3c_g  = (const float*)d_in[11];
  const float* bn3c_b  = (const float*)d_in[12];
  const float* se1_w1  = (const float*)d_in[13];
  const float* se1_w2  = (const float*)d_in[14];
  const float* se2_w1  = (const float*)d_in[15];
  const float* se2_w2  = (const float*)d_in[16];
  const float* se3_w1  = (const float*)d_in[17];
  const float* se3_w2  = (const float*)d_in[18];
  const float* att_w1  = (const float*)d_in[19];
  const float* att_w2  = (const float*)d_in[20];
  const float* fc_w    = (const float*)d_in[21];
  // d_in[22] = fc_b: cancelled exactly by the following BatchNorm1d mean subtraction
  const float* bn2_g   = (const float*)d_in[23];
  const float* bn2_b   = (const float*)d_in[24];
  const float* bias_b  = (const float*)d_in[25];
  const int*   sub     = (const int*)d_in[26];
  const int*   rel     = (const int*)d_in[27];

  char* ws = (char*)d_ws;
  float*  stats = (float*)(ws + OFF_STATS);
  float*  xpre  = (float*)(ws + OFF_XPRE);
  float*  pool  = (float*)(ws + OFF_POOL);
  float*  chsc  = (float*)(ws + OFF_CHSC);
  float*  pA    = (float*)(ws + OFF_PA);
  float*  pB    = (float*)(ws + OFF_PB);
  __bf16* Am    = (__bf16*)(ws + OFF_A);
  __bf16* Wpk   = (__bf16*)(ws + OFF_WPK);
  float*  outfc = (float*)(ws + OFF_OFC);
  __bf16* Hbf   = (__bf16*)(ws + OFF_HBF);
  __bf16* Npk   = (__bf16*)(ws + OFF_NFPK);
  float*  out   = (float*)d_out;

  k_zero_f32<<<4, 256, 0, stream>>>(stats, 1024);
  k_zero_f32<<<(kB * kNFC) / 256, 256, 0, stream>>>(outfc, kB * kNFC);  // FC atomic accumulators
  k_gather<<<kB, 128, 0, stream>>>(n_feats, r_feats, sub, rel, xpre, stats);
  k_conv_stats<<<kB * 3, 256, 0, stream>>>(xpre, stats, pool, filt1, filt2, filt3, rel, bn0_g, bn0_b);
  k_chfin<<<2, 256, 0, stream>>>(stats, bn1_g, bn1_b, bn2c_g, bn2c_b, bn3c_g, bn3c_b, chsc);
  k_gates<<<kB, 128, 0, stream>>>(pool, chsc, pA, pB, se1_w1, se1_w2, se2_w1, se2_w2,
                                  se3_w1, se3_w2, att_w1, att_w2);
  k_buildA<<<kB * 3, 256, 0, stream>>>(xpre, stats, filt1, filt2, filt3, rel, bn0_g, bn0_b, pA, pB, Am);
  k_pack_fcw<<<(kKTfc * kKTfc > 0 ? (kKTfc * kNTfc * 512) / 256 : 1), 256, 0, stream>>>(fc_w, Wpk);
  k_fc_gemm<<<(64 * kNG * kKSpl) / 4, 128, 0, stream>>>(Am, Wpk, outfc);  // 2048 waves, A read 2x
  k_bn2<<<kKH, 256, 0, stream>>>(outfc, bn2_g, bn2_b, Hbf);
  k_pack_nf<<<(kKTsc * kNTsc * 512) / 256, 256, 0, stream>>>(n_feats, Npk);
  k_score_gemm<<<(64 * kNTsc) / 4, 128, 0, stream>>>(Hbf, Npk, bias_b, out); // 160000 wave-tiles
}
